// GQLModel_21449066676259
// MI455X (gfx1250) — compile-verified, matrix operand order checked by reading, not run on hardware
//
#include <hip/hip_runtime.h>
#include <hip/hip_bf16.h>

typedef __bf16 bf16_t;
typedef __attribute__((ext_vector_type(16))) __bf16 v16bf;
typedef __attribute__((ext_vector_type(8)))  __bf16 v8bf;
typedef __attribute__((ext_vector_type(8)))  float  v8f;

#define T_TOT 2048
#define BATCH 1024
#define NA    4
#define DDIM  128
#define CHUNK 16
#define ROWS  (CHUNK * NA)   // 64 GEMM rows per chunk
#define HS    136            // padded H_s row stride (bf16), 272B: 16B-aligned, bank-spread
#define QS    132            // padded Q_s row stride (f32), 528B: 16B-aligned, bank-spread

static __device__ __forceinline__ bf16_t f2bf(float x) {
    unsigned u = __builtin_bit_cast(unsigned, x);
    unsigned r = (u + 0x7FFFu + ((u >> 16) & 1u)) >> 16;   // round-to-nearest-even
    return __builtin_bit_cast(bf16_t, (unsigned short)r);
}
static __device__ __forceinline__ float bf2f(bf16_t h) {
    unsigned u = ((unsigned)__builtin_bit_cast(unsigned short, h)) << 16;
    return __builtin_bit_cast(float, u);
}

__global__ __launch_bounds__(256)
void gql_wmma_kernel(const float* __restrict__ inp,       // (T,1,B,9)
                     const float* __restrict__ phi_raw,   // (P,D)
                     const float* __restrict__ chi_raw,   // (P,D)
                     const float* __restrict__ beta_raw,  // (P,D)
                     const float* __restrict__ kappa_raw, // (P,D)
                     const float* __restrict__ C_raw,     // (P,D,D)
                     float* __restrict__ out)             // logits(T,1,B,A) ++ q(B,A,D) ++ h(B,A,D)
{
    __shared__ __align__(16) bf16_t Ct_s[DDIM * DDIM];  // 32 KB, Ct_s[e*128+d] = clip(C[d][e])
    __shared__ __align__(16) bf16_t H_s[ROWS * HS];     // 17 KB (padded rows)
    __shared__ __align__(16) float  Q_s[ROWS * QS];     // 33 KB (padded rows)
    __shared__ float in_s[CHUNK][8];
    __shared__ float logit_s[ROWS];

    const int b    = blockIdx.x;
    const int tid  = threadIdx.x;
    const int lane = tid & 31;
    const int wave = tid >> 5;

    const int pid = (int)inp[(size_t)b * 9 + 8];        // x[0, b, -1]

    // ---- stage clipped, transposed C[pid] as bf16 into LDS -------------------
    const float* Cp = C_raw + (size_t)pid * DDIM * DDIM;
    for (int i = tid; i < DDIM * DDIM; i += 256) {
        int d = i >> 7, e = i & 127;
        float c = Cp[i];
        c = fminf(fmaxf(c, -10.f), 10.f);
        Ct_s[e * DDIM + d] = f2bf(c);
    }
    __syncthreads();

    // ---- wave owns e-tile [w*16, w*16+16): resident A-fragments of C^T -------
    // WMMA A layout (16-bit 16x32): lanes 0-15: M=lane, K {k0..k0+7, k0+16..k0+23};
    //                               lanes16-31: M=lane-16, K offset +8.
    const int eRowA = wave * 16 + (lane & 15);          // A-matrix row (= e)
    const int kOff  = (lane >> 4) ? 8 : 0;
    v16bf Afrag[4];
#pragma unroll
    for (int kt = 0; kt < 4; ++kt) {
        int k0 = kt * 32 + kOff;
        v8bf lo = *(const v8bf*)&Ct_s[eRowA * DDIM + k0];
        v8bf hi = *(const v8bf*)&Ct_s[eRowA * DDIM + k0 + 16];
        Afrag[kt] = __builtin_shufflevector(lo, hi, 0,1,2,3,4,5,6,7,8,9,10,11,12,13,14,15);
    }

    // ---- epilogue constants: e = e_base + r on the VGPR axis -----------------
    const int e_base = wave * 16 + kOff;                // lanes>=16 cover e_base = +8
    float betav[8], kappav[8];
#pragma unroll
    for (int r = 0; r < 8; ++r) {
        float br = beta_raw[pid * DDIM + e_base + r];
        float sp = (br > 20.f) ? br : log1pf(__expf(br));       // softplus
        betav[r]  = fminf(fmaxf(sp, 0.1f), 10.f);
        float kr = kappa_raw[pid * DDIM + e_base + r];
        kappav[r] = fminf(fmaxf(kr, -10.f), 10.f);
    }

    // ---- per-thread recurrence params: d = tid&127, actions a0, a0+2 ---------
    const int d_idx = tid & 127;
    const int a0 = tid >> 7;        // 0 or 1
    const int a1 = a0 + 2;          // 2 or 3
    float phi, chi;
    {
        float pr = phi_raw[pid * DDIM + d_idx];
        phi = fminf(fmaxf(1.f / (1.f + __expf(-pr)), 0.01f), 0.99f);
        float cr = chi_raw[pid * DDIM + d_idx];
        chi = fminf(fmaxf(1.f / (1.f + __expf(-cr)), 0.01f), 0.99f);
    }
    float q0s = 0.5f, q1s = 0.5f, h0s = 0.f, h1s = 0.f;

    // ---- main loop over 128 chunks of 16 timesteps ---------------------------
    for (int t0 = 0; t0 < T_TOT; t0 += CHUNK) {
        if (tid < CHUNK * 8) {                          // stage actions+rewards
            int tl = tid >> 3, c = tid & 7;
            float v = inp[((size_t)(t0 + tl) * BATCH + b) * 9 + c];
            in_s[tl][c] = (v == v) ? v : 0.f;           // nan_to_num
        }
        if (tid < ROWS) logit_s[tid] = 0.f;
        if (t0 + CHUNK < T_TOT && tid < CHUNK)          // global_prefetch next chunk
            __builtin_prefetch(&inp[((size_t)(t0 + CHUNK + tid) * BATCH + b) * 9], 0, 1);
        __syncthreads();

        // f32 recurrence, 16 steps; each thread owns (a0,d) and (a1,d)
#pragma unroll 4
        for (int tl = 0; tl < CHUNK; ++tl) {
            float act0 = in_s[tl][a0],     rew0 = in_s[tl][4 + a0];
            float act1 = in_s[tl][a1],     rew1 = in_s[tl][4 + a1];
            q0s = (1.f - phi) * q0s + phi * (rew0 * act0);
            h0s = (1.f - chi) * h0s + chi * act0;
            q1s = (1.f - phi) * q1s + phi * (rew1 * act1);
            h1s = (1.f - chi) * h1s + chi * act1;
            int row0 = tl * NA + a0, row1 = tl * NA + a1;
            H_s[row0 * HS + d_idx] = f2bf(h0s);
            H_s[row1 * HS + d_idx] = f2bf(h1s);
            Q_s[row0 * QS + d_idx] = q0s;
            Q_s[row1 * QS + d_idx] = q1s;
        }
        __syncthreads();

        // GEMM (transposed): D[e][row] = sum_d Ct[e,d] * H[row,d]
        // M-dim = e (wave-resident A), N-dim = row (4 tiles, B from LDS H_s)
#pragma unroll
        for (int mt = 0; mt < 4; ++mt) {
            const int row = mt * 16 + (lane & 15);      // B layout: N = lane&15
            v8f acc = {};
#pragma unroll
            for (int kt = 0; kt < 4; ++kt) {
                int k0 = kt * 32 + kOff;
                v8bf lo = *(const v8bf*)&H_s[row * HS + k0];
                v8bf hi = *(const v8bf*)&H_s[row * HS + k0 + 16];
                v16bf Bfrag = __builtin_shufflevector(lo, hi,
                    0,1,2,3,4,5,6,7,8,9,10,11,12,13,14,15);
                acc = __builtin_amdgcn_wmma_f32_16x16x32_bf16(
                    false, Afrag[kt], false, Bfrag, (short)0, acc, false, false);
            }
            // val = sum_{r=0..7} (D[e_base+r][row] + beta)*q[row,e] + kappa*h[row,e]
            float4 qlo = *(const float4*)&Q_s[row * QS + e_base];
            float4 qhi = *(const float4*)&Q_s[row * QS + e_base + 4];
            v8bf  hv  = *(const v8bf*)  &H_s[row * HS + e_base];
            float qv[8] = {qlo.x, qlo.y, qlo.z, qlo.w, qhi.x, qhi.y, qhi.z, qhi.w};
            float val = 0.f;
#pragma unroll
            for (int r = 0; r < 8; ++r)
                val += (acc[r] + betav[r]) * qv[r] + kappav[r] * bf2f(hv[r]);
            val += __shfl_xor(val, 16, 32);             // combine e halves (one bpermute)
            if (lane < 16) atomicAdd(&logit_s[row], val);   // wave-wide ds_add_f32
        }
        __syncthreads();

        if (tid < ROWS) {                               // write logits chunk
            int tl = tid >> 2, a = tid & 3;
            out[((size_t)(t0 + tl) * BATCH + b) * NA + a] = logit_s[tid];
        }
        __syncthreads();
    }

    // ---- final q, h (exact f32 recurrence state) -----------------------------
    {
        size_t qbase = (size_t)T_TOT * BATCH * NA;
        size_t hbase = qbase + (size_t)BATCH * NA * DDIM;
        out[qbase + ((size_t)b * NA + a0) * DDIM + d_idx] = q0s;
        out[qbase + ((size_t)b * NA + a1) * DDIM + d_idx] = q1s;
        out[hbase + ((size_t)b * NA + a0) * DDIM + d_idx] = h0s;
        out[hbase + ((size_t)b * NA + a1) * DDIM + d_idx] = h1s;
    }
}

extern "C" void kernel_launch(void* const* d_in, const int* in_sizes, int n_in,
                              void* d_out, int out_size, void* d_ws, size_t ws_size,
                              hipStream_t stream) {
    const float* inp       = (const float*)d_in[0];
    const float* phi_raw   = (const float*)d_in[1];
    const float* chi_raw   = (const float*)d_in[2];
    const float* beta_raw  = (const float*)d_in[3];
    const float* kappa_raw = (const float*)d_in[4];
    const float* C_raw     = (const float*)d_in[5];
    float* out = (float*)d_out;
    (void)in_sizes; (void)n_in; (void)out_size; (void)d_ws; (void)ws_size;

    gql_wmma_kernel<<<dim3(BATCH), dim3(256), 0, stream>>>(
        inp, phi_raw, chi_raw, beta_raw, kappa_raw, C_raw, out);
}
